// PhiAttention_24721831756203
// MI455X (gfx1250) — compile-verified, matrix operand order checked by reading, not run on hardware
//
#include <hip/hip_runtime.h>
#include <hip/hip_bf16.h>

// ---------------------------------------------------------------------------
// Phi attention block on gfx1250 (MI455X): bf16 WMMA GEMMs + flash attention,
// with CDNA5 async global->LDS staging (ASYNCcnt) for the GEMM/Q/K tiles.
// ---------------------------------------------------------------------------

typedef __attribute__((ext_vector_type(16))) __bf16 v16bf;
typedef __attribute__((ext_vector_type(8)))  float  v8f;

#define EMBED 2560
#define HEADS 8
#define HDIM  320                 // EMBED / HEADS
#define BATCH 2
#define SEQ   2048
#define M_TOT (BATCH * SEQ)       // 4096
#define QKVN  (3 * EMBED)         // 7680

__device__ __forceinline__ unsigned short f32_bf16(float f) {
    unsigned int u = __float_as_uint(f);
    u += 0x7FFFu + ((u >> 16) & 1u);          // round-to-nearest-even
    return (unsigned short)(u >> 16);
}

// CDNA5 async copy: GLOBAL_LOAD_ASYNC_TO_LDS_B128 (ASYNCcnt-tracked, no VGPR
// data path). VDST VGPR carries the workgroup-relative LDS byte offset; for
// LDS the generic pointer's low 32 bits are exactly that offset.
__device__ __forceinline__ void async_g2l_b128(void* lds_dst, const void* src) {
    unsigned ldsoff = (unsigned)(unsigned long long)lds_dst;
    asm volatile("global_load_async_to_lds_b128 %0, %1, off"
                 :: "v"(ldsoff), "v"(src) : "memory");
}
__device__ __forceinline__ void wait_async0() {
    asm volatile("s_wait_asynccnt 0" ::: "memory");
}

union Frag16 { v16bf v; uint4 q[2]; };

// Load one 16x32 bf16 A-fragment (MxK) -- or a 32x16 B-fragment read as the
// [n][k] K-contiguous layout -- from LDS with row stride ldk (elements).
// CDNA5 layout (ISA 7.12.2): lane L holds row (L&15); elements 0..7 cover
// K = 8*(L>>4)+j, elements 8..15 cover K = 16 + 8*(L>>4)+j.
// Two 16B ds reads per fragment.
__device__ __forceinline__ v16bf load_frag_ds(const __bf16* base, int ldk) {
    const int lane = threadIdx.x & 31;
    const __bf16* p = base + (lane & 15) * ldk + ((lane >> 4) << 3);
    Frag16 f;
    f.q[0] = *(const uint4*)(p);
    f.q[1] = *(const uint4*)(p + 16);
    return f.v;
}

__device__ __forceinline__ v8f wmma_bf16(v16bf a, v16bf b, v8f c) {
    // (neg_a, A, neg_b, B, c_mod, C, reuse_a, reuse_b)
    return __builtin_amdgcn_wmma_f32_16x16x32_bf16(false, a, false, b,
                                                   (short)0, c, false, false);
}

// ---------------------------------------------------------------------------
// fp32 -> bf16 conversion (memory bound, one pass)
// ---------------------------------------------------------------------------
__global__ void cvt_f32_to_bf16(const float* __restrict__ in,
                                unsigned short* __restrict__ out, int n) {
    int i = blockIdx.x * blockDim.x + threadIdx.x;
    int stride = gridDim.x * blockDim.x;
    for (; i < n; i += stride) out[i] = f32_bf16(in[i]);
}

// ---------------------------------------------------------------------------
// GEMM1: qkv[m,f] = sum_e x[m,e] * Wqkv[f,e] + b[f], scattered to
// q/k/v bf16 buffers laid out [B, H, S, D] (one contiguous qkv buffer).
// Tile 128x128x32, 256 threads = 8 waves, wave tile 32x64 (2x4 WMMA tiles).
// ---------------------------------------------------------------------------
__global__ __launch_bounds__(256)
void gemm_qkv_wmma(const __bf16* __restrict__ A,     // [M_TOT][EMBED]
                   const __bf16* __restrict__ W,     // [QKVN][EMBED]
                   const float*  __restrict__ bias,  // [QKVN]
                   __bf16* __restrict__ qkv)         // 3 x [B,H,S,D]
{
    __shared__ __align__(16) __bf16 lA[128 * 32];
    __shared__ __align__(16) __bf16 lB[128 * 32];

    const int tid  = threadIdx.x;
    const int wave = tid >> 5;        // wave32
    const int lane = tid & 31;
    const int m0 = blockIdx.y * 128;
    const int n0 = blockIdx.x * 128;
    const int wr = (wave & 3) * 32;   // wave row base inside tile
    const int wc = (wave >> 2) * 64;  // wave col base inside tile

    v8f acc[2][4];
    for (int i = 0; i < 2; i++)
        for (int j = 0; j < 4; j++)
            acc[i][j] = (v8f){0, 0, 0, 0, 0, 0, 0, 0};

    const int c0 = tid * 2;           // staging: 512 uint4 per tile, 2/thread
    for (int k0 = 0; k0 < EMBED; k0 += 32) {
        // Async global->LDS staging (no VGPR data path, ASYNCcnt-tracked).
#pragma unroll
        for (int i = 0; i < 2; i++) {
            int c   = c0 + i;
            int row = c >> 2;
            int kc  = (c & 3) << 3;
            async_g2l_b128(&lA[row * 32 + kc],
                           &A[(size_t)(m0 + row) * EMBED + k0 + kc]);
            async_g2l_b128(&lB[row * 32 + kc],
                           &W[(size_t)(n0 + row) * EMBED + k0 + kc]);
        }
        if (k0 + 32 < EMBED) {        // global_prefetch_b8 of next K tile
            __builtin_prefetch(&A[(size_t)(m0 + (tid >> 1)) * EMBED + k0 + 32], 0, 1);
            __builtin_prefetch(&W[(size_t)(n0 + (tid >> 1)) * EMBED + k0 + 32], 0, 1);
        }
        wait_async0();
        __syncthreads();

        v16bf af[2], bf[4];
#pragma unroll
        for (int i = 0; i < 2; i++) af[i] = load_frag_ds(&lA[(wr + i * 16) * 32], 32);
#pragma unroll
        for (int j = 0; j < 4; j++) bf[j] = load_frag_ds(&lB[(wc + j * 16) * 32], 32);
#pragma unroll
        for (int i = 0; i < 2; i++)
#pragma unroll
            for (int j = 0; j < 4; j++)
                acc[i][j] = wmma_bf16(af[i], bf[j], acc[i][j]);
        __syncthreads();
    }

    // Epilogue: bias add + scatter into [sel][B][H][S][D] bf16.
    const int half = lane >> 4;
    const int ncol = lane & 15;
    const size_t selStride = (size_t)M_TOT * EMBED;   // B*H*S*D elements
#pragma unroll
    for (int i = 0; i < 2; i++) {
#pragma unroll
        for (int j = 0; j < 4; j++) {
            int col = n0 + wc + j * 16 + ncol;        // f index in [0, 3E)
            float bv  = bias[col];
            int sel = col / EMBED;
            int rem = col - sel * EMBED;
            int h   = rem / HDIM;
            int d   = rem - h * HDIM;
            unsigned short* dst = (unsigned short*)(qkv + sel * selStride);
#pragma unroll
            for (int r = 0; r < 8; r++) {
                int row = m0 + wr + i * 16 + half * 8 + r;   // m in [0, 4096)
                int b   = row >> 11;                          // / SEQ
                int s   = row & (SEQ - 1);
                dst[(((size_t)b * HEADS + h) * SEQ + s) * HDIM + d] =
                    f32_bf16(acc[i][j][r] + bv);
            }
        }
    }
}

// ---------------------------------------------------------------------------
// Flash attention: one block per (b, h, 32-row q tile); stream 32-key blocks.
// 256 threads = 8 waves. O accumulators: wave owns 16 q-rows x 80 d-cols
// (5 WMMA tiles). Online softmax state (m, l) kept in LDS per q row.
// ---------------------------------------------------------------------------
__global__ __launch_bounds__(256)
void flash_attn_wmma(const __bf16* __restrict__ Q,   // [B,H,S,D]
                     const __bf16* __restrict__ K,
                     const __bf16* __restrict__ V,
                     __bf16* __restrict__ O)         // [B,S,E]
{
    __shared__ __align__(16) __bf16 lQ[32 * HDIM];   // [qrow][d]
    __shared__ __align__(16) __bf16 lK[32 * HDIM];   // [key][d]
    __shared__ __align__(16) __bf16 lVT[HDIM * 32];  // [d][key]  (transposed)
    __shared__ __align__(16) float  lS[32 * 32];     // scores fp32
    __shared__ __align__(16) __bf16 lP[32 * 32];     // probs bf16 [q][key]
    __shared__ float lM[32], lL[32], lAlpha[32];

    const int tid  = threadIdx.x;
    const int wave = tid >> 5;
    const int lane = tid & 31;
    const int q0 = blockIdx.x * 32;
    const int h  = blockIdx.y;
    const int b  = blockIdx.z;
    const size_t headBase = ((size_t)b * HEADS + h) * (size_t)SEQ * HDIM;

    // Stage Q tile once (async): 32*320 bf16 = 1280 x b128, 5 per thread.
#pragma unroll
    for (int i = 0; i < 5; i++) {
        int c   = tid + 256 * i;
        int row = c / 40;
        int kc  = (c % 40) << 3;
        async_g2l_b128(&lQ[row * HDIM + kc],
                       &Q[headBase + (size_t)(q0 + row) * HDIM + kc]);
    }
    if (tid < 32) { lM[tid] = -3.0e38f; lL[tid] = 0.0f; }

    const int qr = wave & 1;                  // 16-row O tile
    const int cg = wave >> 1;                 // col group: 5 tiles (80 cols)
    v8f acc[5];
#pragma unroll
    for (int i = 0; i < 5; i++) acc[i] = (v8f){0, 0, 0, 0, 0, 0, 0, 0};
    const float scale = 0.05590169943749474f; // 1/sqrt(320)

    wait_async0();
    __syncthreads();

    for (int kb = 0; kb < SEQ; kb += 32) {
        // Stage K block [key][d] (async).
#pragma unroll
        for (int i = 0; i < 5; i++) {
            int c   = tid + 256 * i;
            int row = c / 40;
            int kc  = (c % 40) << 3;
            async_g2l_b128(&lK[row * HDIM + kc],
                           &K[headBase + (size_t)(kb + row) * HDIM + kc]);
        }
        // Stage V transposed: LDS [d][key] so PV B-frags are K-contiguous
        // (register path: needs the transpose scatter).
#pragma unroll
        for (int i = 0; i < 5; i++) {
            int c   = tid + 256 * i;
            int row = c / 40;
            int d0  = (c % 40) << 3;
            uint4 pkt = *(const uint4*)&V[headBase + (size_t)(kb + row) * HDIM + d0];
            const unsigned short* e = (const unsigned short*)&pkt;
#pragma unroll
            for (int j = 0; j < 8; j++)
                ((unsigned short*)lVT)[(d0 + j) * 32 + row] = e[j];
        }
        wait_async0();
        __syncthreads();

        // S = (Q K^T) * scale  -- 4 WMMA tiles on waves 0..3, K=320 in 10 steps
        if (wave < 4) {
            int sr = (wave & 1) * 16;         // q row tile
            int sc = (wave >> 1) * 16;        // key col tile
            v8f s = (v8f){0, 0, 0, 0, 0, 0, 0, 0};
#pragma unroll
            for (int kk = 0; kk < HDIM; kk += 32) {
                v16bf a  = load_frag_ds(&lQ[sr * HDIM + kk], HDIM);
                v16bf bb = load_frag_ds(&lK[sc * HDIM + kk], HDIM);
                s = wmma_bf16(a, bb, s);
            }
            int col = sc + (lane & 15);
            int rb  = sr + (lane >> 4) * 8;
#pragma unroll
            for (int r = 0; r < 8; r++) lS[(rb + r) * 32 + col] = s[r] * scale;
        }
        __syncthreads();

        // Online softmax, one q-row per thread 0..31.
        if (tid < 32) {
            float mold = lM[tid];
            float mnew = mold;
            for (int c = 0; c < 32; c++) mnew = fmaxf(mnew, lS[tid * 32 + c]);
            float alpha = __expf(mold - mnew);
            float lsum  = lL[tid] * alpha;
            for (int c = 0; c < 32; c++) {
                float p = __expf(lS[tid * 32 + c] - mnew);
                lsum += p;
                ((unsigned short*)lP)[tid * 32 + c] = f32_bf16(p);
            }
            lM[tid] = mnew; lL[tid] = lsum; lAlpha[tid] = alpha;
        }
        __syncthreads();

        // O = O*alpha + P @ V   (all 8 waves; K=32 = one WMMA step)
        {
            int rb = qr * 16 + (lane >> 4) * 8;
            float al[8];
#pragma unroll
            for (int r = 0; r < 8; r++) al[r] = lAlpha[rb + r];
            v16bf a = load_frag_ds(&lP[qr * 16 * 32], 32);
#pragma unroll
            for (int i = 0; i < 5; i++) {
#pragma unroll
                for (int r = 0; r < 8; r++) acc[i][r] *= al[r];
                v16bf bb = load_frag_ds(&lVT[(cg * 80 + i * 16) * 32], 32);
                acc[i] = wmma_bf16(a, bb, acc[i]);
            }
        }
        __syncthreads();
    }

    // Final 1/l normalization; write attn out bf16 at [b, q0+row, h*D + d].
    const int rb = qr * 16 + (lane >> 4) * 8;
#pragma unroll
    for (int i = 0; i < 5; i++) {
        int d = cg * 80 + i * 16 + (lane & 15);
#pragma unroll
        for (int r = 0; r < 8; r++) {
            int row = rb + r;
            float o = acc[i][r] / lL[row];
            ((unsigned short*)O)[((size_t)b * SEQ + q0 + row) * EMBED + h * HDIM + d] =
                f32_bf16(o);
        }
    }
}

// ---------------------------------------------------------------------------
// GEMM2: out[m,f] = sum_e attn[m,e] * Wout[f,e] + b[f]  (fp32 output)
// ---------------------------------------------------------------------------
__global__ __launch_bounds__(256)
void gemm_out_wmma(const __bf16* __restrict__ A,     // [M_TOT][EMBED]
                   const __bf16* __restrict__ W,     // [EMBED][EMBED]
                   const float*  __restrict__ bias,  // [EMBED]
                   float* __restrict__ out)          // [M_TOT][EMBED]
{
    __shared__ __align__(16) __bf16 lA[128 * 32];
    __shared__ __align__(16) __bf16 lB[128 * 32];

    const int tid  = threadIdx.x;
    const int wave = tid >> 5;
    const int lane = tid & 31;
    const int m0 = blockIdx.y * 128;
    const int n0 = blockIdx.x * 128;
    const int wr = (wave & 3) * 32;
    const int wc = (wave >> 2) * 64;

    v8f acc[2][4];
    for (int i = 0; i < 2; i++)
        for (int j = 0; j < 4; j++)
            acc[i][j] = (v8f){0, 0, 0, 0, 0, 0, 0, 0};

    const int c0 = tid * 2;
    for (int k0 = 0; k0 < EMBED; k0 += 32) {
#pragma unroll
        for (int i = 0; i < 2; i++) {
            int c   = c0 + i;
            int row = c >> 2;
            int kc  = (c & 3) << 3;
            async_g2l_b128(&lA[row * 32 + kc],
                           &A[(size_t)(m0 + row) * EMBED + k0 + kc]);
            async_g2l_b128(&lB[row * 32 + kc],
                           &W[(size_t)(n0 + row) * EMBED + k0 + kc]);
        }
        if (k0 + 32 < EMBED) {
            __builtin_prefetch(&A[(size_t)(m0 + (tid >> 1)) * EMBED + k0 + 32], 0, 1);
            __builtin_prefetch(&W[(size_t)(n0 + (tid >> 1)) * EMBED + k0 + 32], 0, 1);
        }
        wait_async0();
        __syncthreads();

        v16bf af[2], bf[4];
#pragma unroll
        for (int i = 0; i < 2; i++) af[i] = load_frag_ds(&lA[(wr + i * 16) * 32], 32);
#pragma unroll
        for (int j = 0; j < 4; j++) bf[j] = load_frag_ds(&lB[(wc + j * 16) * 32], 32);
#pragma unroll
        for (int i = 0; i < 2; i++)
#pragma unroll
            for (int j = 0; j < 4; j++)
                acc[i][j] = wmma_bf16(af[i], bf[j], acc[i][j]);
        __syncthreads();
    }

    const int half = lane >> 4;
    const int ncol = lane & 15;
#pragma unroll
    for (int i = 0; i < 2; i++) {
#pragma unroll
        for (int j = 0; j < 4; j++) {
            int col  = n0 + wc + j * 16 + ncol;
            float bv = bias[col];
#pragma unroll
            for (int r = 0; r < 8; r++) {
                int row = m0 + wr + i * 16 + half * 8 + r;
                out[(size_t)row * EMBED + col] = acc[i][j][r] + bv;
            }
        }
    }
}

// ---------------------------------------------------------------------------
extern "C" void kernel_launch(void* const* d_in, const int* in_sizes, int n_in,
                              void* d_out, int out_size, void* d_ws, size_t ws_size,
                              hipStream_t stream) {
    const float* x    = (const float*)d_in[0];  // [2,2048,2560]
    const float* Wqkv = (const float*)d_in[1];  // [7680,2560]
    const float* bqkv = (const float*)d_in[2];  // [7680]
    const float* Wout = (const float*)d_in[3];  // [2560,2560]
    const float* bout = (const float*)d_in[4];  // [2560]
    float* out = (float*)d_out;

    // Workspace layout (bf16 regions, all 16B aligned).
    char* ws = (char*)d_ws;
    __bf16* xb   = (__bf16*)ws; ws += (size_t)M_TOT * EMBED * 2;   // 20.97 MB
    __bf16* wqb  = (__bf16*)ws; ws += (size_t)QKVN  * EMBED * 2;   // 39.32 MB
    __bf16* wob  = (__bf16*)ws; ws += (size_t)EMBED * EMBED * 2;   // 13.11 MB
    __bf16* qkvb = (__bf16*)ws; ws += 3 * (size_t)M_TOT * EMBED * 2; // 62.9 MB
    __bf16* attn = (__bf16*)ws; ws += (size_t)M_TOT * EMBED * 2;   // 20.97 MB
    (void)ws_size; (void)in_sizes; (void)n_in; (void)out_size;

    const size_t headTensor = (size_t)M_TOT * EMBED;  // == B*H*S*D elements

    // 1) fp32 -> bf16 conversions
    cvt_f32_to_bf16<<<2048, 256, 0, stream>>>(x,    (unsigned short*)xb,  M_TOT * EMBED);
    cvt_f32_to_bf16<<<2048, 256, 0, stream>>>(Wqkv, (unsigned short*)wqb, QKVN  * EMBED);
    cvt_f32_to_bf16<<<2048, 256, 0, stream>>>(Wout, (unsigned short*)wob, EMBED * EMBED);

    // 2) QKV projection (WMMA) with [B,H,S,D] scatter
    gemm_qkv_wmma<<<dim3(QKVN / 128, M_TOT / 128), 256, 0, stream>>>(
        xb, wqb, bqkv, qkvb);

    // 3) Flash attention (WMMA QK^T and PV, online softmax)
    flash_attn_wmma<<<dim3(SEQ / 32, HEADS, BATCH), 256, 0, stream>>>(
        qkvb, qkvb + headTensor, qkvb + 2 * headTensor, attn);

    // 4) Output projection (WMMA) -> fp32 + bias
    gemm_out_wmma<<<dim3(EMBED / 128, M_TOT / 128), 256, 0, stream>>>(
        attn, wob, bout, out);
}
